// GCN_Attention_86887188398715
// MI455X (gfx1250) — matrix operand order repaired and checked by reading, not compiled, not run on hardware
//
#include <hip/hip_runtime.h>
#include <math.h>

#define NFEAT  512
#define NHID   256
#define NCLASS 16
#define NSTRUC 64

typedef __attribute__((ext_vector_type(16))) __bf16 v16bf;
typedef __attribute__((ext_vector_type(8)))  __bf16 v8bf;
typedef __attribute__((ext_vector_type(8)))  float  v8f;

static __device__ __forceinline__ v8f wmma_bf16(v16bf a, v16bf b, v8f c) {
    // D = A(16x32 bf16) * B(32x16 bf16) + C(16x16 f32)
    return __builtin_amdgcn_wmma_f32_16x16x32_bf16(
        /*neg_a=*/false, a, /*neg_b=*/false, b,
        /*c_mod=*/(short)0, c, /*reuse_a=*/false, /*reuse_b=*/false);
}

// A fragment (16x32 bf16, rows m0..m0+15, K window [k0,k0+32)) from row-major bf16.
// Lane L: row m=L&15; K runs [8*(L>>4), +8) and [16+8*(L>>4), +8)  (ISA 7.12.2)
static __device__ __forceinline__ v16bf load_a_frag(const __bf16* __restrict__ src,
                                                    int m0, int ldk, int k0, int lane) {
    const int lm = lane & 15, hgrp = lane >> 4;
    const __bf16* ap = src + (size_t)(m0 + lm) * ldk + k0 + 8 * hgrp;
    v8bf lo = *(const v8bf*)(ap);        // 16B load
    v8bf hi = *(const v8bf*)(ap + 16);   // 16B load
    return __builtin_shufflevector(lo, hi, 0,1,2,3,4,5,6,7,8,9,10,11,12,13,14,15);
}

// B fragment (32x16 bf16): lane L holds column n=L&15; K runs [8*(L>>4), +8) and [16+8*(L>>4), +8)
static __device__ __forceinline__ v16bf load_b_frag(const __bf16* __restrict__ src,
                                                    int ldn, int k0, int n0, int lane) {
    const int lm = lane & 15, hgrp = lane >> 4;
    v16bf b;
#pragma unroll
    for (int s = 0; s < 16; ++s) {
        const int kk = (s < 8) ? (8 * hgrp + s) : (16 + 8 * hgrp + (s - 8));
        b[s] = src[(size_t)(k0 + kk) * ldn + n0 + lm];
    }
    return b;
}

// ---------------- GEMM1: xw[N,256] = xbf[N,512] @ W1bf[512,256] ----------------
__global__ __launch_bounds__(256) void k_gemm1(const __bf16* __restrict__ xbf,
                                               const __bf16* __restrict__ W1bf,
                                               float* __restrict__ xw,
                                               int n_row_tiles) {
    const int lane  = threadIdx.x & 31;
    const int gwave = (int)((blockIdx.x * blockDim.x + threadIdx.x) >> 5);
    const int row_tile = gwave >> 2;   // 4 slabs of 64 columns
    const int slab     = gwave & 3;
    if (row_tile >= n_row_tiles) return;
    const int m0 = row_tile * 16;
    const int n0 = slab * 64;
    const int lm = lane & 15, hgrp = lane >> 4;

    v8f acc[4] = {v8f{}, v8f{}, v8f{}, v8f{}};
    for (int k0 = 0; k0 < NFEAT; k0 += 32) {
        v16bf a = load_a_frag(xbf, m0, NFEAT, k0, lane);
#pragma unroll
        for (int t = 0; t < 4; ++t) {
            v16bf b = load_b_frag(W1bf, NHID, k0, n0 + t * 16, lane);
            acc[t] = wmma_bf16(a, b, acc[t]);
        }
    }
#pragma unroll
    for (int t = 0; t < 4; ++t) {
        const int nn = n0 + t * 16 + lm;
#pragma unroll
        for (int v = 0; v < 8; ++v)
            xw[(size_t)(m0 + v + 8 * hgrp) * NHID + nn] = acc[t][v];
    }
}

// ---------------- GEMM2: lp[N,16] = hbf[N,256] @ W2bf[256,16] ----------------
__global__ __launch_bounds__(256) void k_gemm2(const __bf16* __restrict__ hbf,
                                               const __bf16* __restrict__ W2bf,
                                               float* __restrict__ lp,
                                               int n_row_tiles) {
    const int lane  = threadIdx.x & 31;
    const int gwave = (int)((blockIdx.x * blockDim.x + threadIdx.x) >> 5);
    if (gwave >= n_row_tiles) return;
    const int m0 = gwave * 16;
    const int lm = lane & 15, hgrp = lane >> 4;

    v8f acc = {};
    for (int k0 = 0; k0 < NHID; k0 += 32) {
        v16bf a = load_a_frag(hbf, m0, NHID, k0, lane);
        v16bf b = load_b_frag(W2bf, NCLASS, k0, 0, lane);
        acc = wmma_bf16(a, b, acc);
    }
#pragma unroll
    for (int v = 0; v < 8; ++v)
        lp[(size_t)(m0 + v + 8 * hgrp) * NCLASS + lm] = acc[v];
}

// -------- Encoder: y[N,64] = sigmoid(hbf @ WeT + be) * att ; WeT is [256,64] bf16 --------
__global__ __launch_bounds__(256) void k_enc(const __bf16* __restrict__ hbf,
                                             const __bf16* __restrict__ WeT,
                                             const float* __restrict__ be,
                                             const float* __restrict__ att,
                                             float* __restrict__ y,
                                             int n_row_tiles) {
    const int lane  = threadIdx.x & 31;
    const int gwave = (int)((blockIdx.x * blockDim.x + threadIdx.x) >> 5);
    if (gwave >= n_row_tiles) return;
    const int m0 = gwave * 16;
    const int lm = lane & 15, hgrp = lane >> 4;

    v8f acc[4] = {v8f{}, v8f{}, v8f{}, v8f{}};
    for (int k0 = 0; k0 < NHID; k0 += 32) {
        v16bf a = load_a_frag(hbf, m0, NHID, k0, lane);
#pragma unroll
        for (int t = 0; t < 4; ++t) {
            v16bf b = load_b_frag(WeT, NSTRUC, k0, t * 16, lane);
            acc[t] = wmma_bf16(a, b, acc[t]);
        }
    }
#pragma unroll
    for (int t = 0; t < 4; ++t) {
        const int nn = t * 16 + lm;
        const float bias = be[nn];
#pragma unroll
        for (int v = 0; v < 8; ++v) {
            const int mm = m0 + v + 8 * hgrp;
            const float z   = acc[t][v] + bias;
            const float sig = 1.0f / (1.0f + __expf(-z));
            y[(size_t)mm * NSTRUC + nn] = sig * att[(size_t)mm * NSTRUC + nn];
        }
    }
}

// ---------------- conversion / helper kernels ----------------
// f32 -> bf16, 8 elements per thread (total must be a multiple of 8)
__global__ void k_cvt8(const float* __restrict__ in, __bf16* __restrict__ out, int total8) {
    const int i = blockIdx.x * blockDim.x + threadIdx.x;
    if (i >= total8) return;
    const float4* p = (const float4*)in + 2 * (size_t)i;
    const float4 a = p[0], b = p[1];
    v8bf o;
    o[0] = (__bf16)a.x; o[1] = (__bf16)a.y; o[2] = (__bf16)a.z; o[3] = (__bf16)a.w;
    o[4] = (__bf16)b.x; o[5] = (__bf16)b.y; o[6] = (__bf16)b.z; o[7] = (__bf16)b.w;
    *(v8bf*)(out + 8 * (size_t)i) = o;
}

// relu(f32) -> bf16, 8 per thread (hacc left pre-relu; only hbf consumed downstream)
__global__ void k_relu_cvt8(const float* __restrict__ in, __bf16* __restrict__ out, int total8) {
    const int i = blockIdx.x * blockDim.x + threadIdx.x;
    if (i >= total8) return;
    const float4* p = (const float4*)in + 2 * (size_t)i;
    float4 a = p[0], b = p[1];
    v8bf o;
    o[0] = (__bf16)fmaxf(a.x, 0.f); o[1] = (__bf16)fmaxf(a.y, 0.f);
    o[2] = (__bf16)fmaxf(a.z, 0.f); o[3] = (__bf16)fmaxf(a.w, 0.f);
    o[4] = (__bf16)fmaxf(b.x, 0.f); o[5] = (__bf16)fmaxf(b.y, 0.f);
    o[6] = (__bf16)fmaxf(b.z, 0.f); o[7] = (__bf16)fmaxf(b.w, 0.f);
    *(v8bf*)(out + 8 * (size_t)i) = o;
}

// We[64,256] f32 -> WeT[256,64] bf16 (transpose); tid = k*64+n -> coalesced writes
__global__ void k_cvt_tr(const float* __restrict__ We, __bf16* __restrict__ WeT, int total) {
    const int i = blockIdx.x * blockDim.x + threadIdx.x;
    if (i >= total) return;
    const int k = i >> 6, n = i & 63;
    WeT[i] = (__bf16)We[(size_t)n * NHID + k];
}

__global__ void k_init_bias(float* __restrict__ dst, const float* __restrict__ bias,
                            int mask, int total) {
    const int i = blockIdx.x * blockDim.x + threadIdx.x;
    if (i < total) dst[i] = bias[i & mask];
}

// scatter-add over edges, 256 features: 64 lanes/edge, float4 each
__global__ void k_scatter1(const int* __restrict__ erow, const int* __restrict__ ecol,
                           const float* __restrict__ ev, const float* __restrict__ xw,
                           float* __restrict__ hacc, int n_edges) {
    const long tid = (long)blockIdx.x * blockDim.x + threadIdx.x;
    const long e   = tid >> 6;
    const int part = (int)(tid & 63);
    if (e >= n_edges) return;
    const int r = erow[e], c = ecol[e];
    const float v = ev[e];
    const float4 s = ((const float4*)(xw + (size_t)c * NHID))[part];
    float* dst = hacc + (size_t)r * NHID + part * 4;
    atomicAdd(dst + 0, v * s.x);
    atomicAdd(dst + 1, v * s.y);
    atomicAdd(dst + 2, v * s.z);
    atomicAdd(dst + 3, v * s.w);
}

// scatter-add over edges, 16 logits: 16 lanes/edge
__global__ void k_scatter2(const int* __restrict__ erow, const int* __restrict__ ecol,
                           const float* __restrict__ ev, const float* __restrict__ lp,
                           float* __restrict__ lacc, int n_edges) {
    const long tid = (long)blockIdx.x * blockDim.x + threadIdx.x;
    const long e   = tid >> 4;
    const int n    = (int)(tid & 15);
    if (e >= n_edges) return;
    const int r = erow[e], c = ecol[e];
    atomicAdd(lacc + (size_t)r * NCLASS + n, ev[e] * lp[(size_t)c * NCLASS + n]);
}

__global__ void k_lsm(const float* __restrict__ lacc, float* __restrict__ out, int n) {
    const int r = blockIdx.x * blockDim.x + threadIdx.x;
    if (r >= n) return;
    float vbuf[NCLASS];
    float mx = -3.402823466e38f;
#pragma unroll
    for (int i = 0; i < NCLASS; ++i) {
        vbuf[i] = lacc[(size_t)r * NCLASS + i];
        mx = fmaxf(mx, vbuf[i]);
    }
    float s = 0.0f;
#pragma unroll
    for (int i = 0; i < NCLASS; ++i) s += __expf(vbuf[i] - mx);
    const float lse = mx + __logf(s);
#pragma unroll
    for (int i = 0; i < NCLASS; ++i) out[(size_t)r * NCLASS + i] = vbuf[i] - lse;
}

extern "C" void kernel_launch(void* const* d_in, const int* in_sizes, int n_in,
                              void* d_out, int out_size, void* d_ws, size_t ws_size,
                              hipStream_t stream) {
    const float* x    = (const float*)d_in[0];
    const int*   erow = (const int*)  d_in[1];
    const int*   ecol = (const int*)  d_in[2];
    const float* ev   = (const float*)d_in[3];
    const float* att  = (const float*)d_in[4];
    const float* W1   = (const float*)d_in[5];
    const float* b1   = (const float*)d_in[6];
    const float* W2   = (const float*)d_in[7];
    const float* b2   = (const float*)d_in[8];
    const float* We   = (const float*)d_in[9];
    const float* be   = (const float*)d_in[10];

    const int n_nodes = in_sizes[0] / NFEAT;
    const int n_edges = in_sizes[1];
    const int row_tiles = (n_nodes + 15) / 16;   // 3125 (exact)

    // workspace layout (16B-aligned sections)
    float*  xw   = (float*)d_ws;                          // [N,256] f32
    float*  hacc = xw   + (size_t)n_nodes * NHID;         // [N,256] f32
    float*  lp   = hacc + (size_t)n_nodes * NHID;         // [N,16]  f32
    float*  lacc = lp   + (size_t)n_nodes * NCLASS;       // [N,16]  f32
    __bf16* xbf  = (__bf16*)(lacc + (size_t)n_nodes * NCLASS); // [N,512] bf16
    __bf16* W1bf = xbf  + (size_t)n_nodes * NFEAT;        // [512,256] bf16
    __bf16* W2bf = W1bf + (size_t)NFEAT * NHID;           // [256,16]  bf16
    __bf16* WeT  = W2bf + (size_t)NHID * NCLASS;          // [256,64]  bf16
    __bf16* hbf  = xbf;                                   // alias: xbf dead after gemm1

    float* out_lsm = (float*)d_out;                       // [N,16]
    float* out_y   = out_lsm + (size_t)n_nodes * NCLASS;  // [N,64]

    // 0) precision pre-pass: one bandwidth-bound conversion instead of per-K-step cvt
    {
        const int t8 = n_nodes * NFEAT / 8;
        k_cvt8<<<(t8 + 255) / 256, 256, 0, stream>>>(x, xbf, t8);
    }
    {
        const int t8 = NFEAT * NHID / 8;
        k_cvt8<<<(t8 + 255) / 256, 256, 0, stream>>>(W1, W1bf, t8);
    }
    {
        const int t8 = NHID * NCLASS / 8;
        k_cvt8<<<(t8 + 255) / 256, 256, 0, stream>>>(W2, W2bf, t8);
    }
    {
        const int t = NHID * NSTRUC;
        k_cvt_tr<<<(t + 255) / 256, 256, 0, stream>>>(We, WeT, t);
    }

    // 1) xw = x @ W1   (bf16 WMMA, pure load+wmma inner loop)
    {
        const int waves  = row_tiles * 4;
        const int blocks = (waves + 7) / 8;
        k_gemm1<<<blocks, 256, 0, stream>>>(xbf, W1bf, xw, row_tiles);
    }
    // 2) hacc = broadcast(b1)
    {
        const int total = n_nodes * NHID;
        k_init_bias<<<(total + 255) / 256, 256, 0, stream>>>(hacc, b1, NHID - 1, total);
    }
    // 3) hacc += scatter(edge_val * xw[col])   (L2-resident fp32 atomics)
    {
        const long threads = (long)n_edges * 64;
        k_scatter1<<<(int)((threads + 255) / 256), 256, 0, stream>>>(erow, ecol, ev, xw, hacc, n_edges);
    }
    // 4) hbf = bf16(relu(hacc))   (fused; hbf reuses xbf region)
    {
        const int t8 = n_nodes * NHID / 8;
        k_relu_cvt8<<<(t8 + 255) / 256, 256, 0, stream>>>(hacc, hbf, t8);
    }
    // 5) lp = h @ W2  (bf16 WMMA)
    k_gemm2<<<(row_tiles + 7) / 8, 256, 0, stream>>>(hbf, W2bf, lp, row_tiles);
    // 6) lacc = broadcast(b2)
    {
        const int total = n_nodes * NCLASS;
        k_init_bias<<<(total + 255) / 256, 256, 0, stream>>>(lacc, b2, NCLASS - 1, total);
    }
    // 7) lacc += scatter(edge_val * lp[col])
    {
        const long threads = (long)n_edges * NCLASS;
        k_scatter2<<<(int)((threads + 255) / 256), 256, 0, stream>>>(erow, ecol, ev, lp, lacc, n_edges);
    }
    // 8) out1 = log_softmax(lacc)
    k_lsm<<<(n_nodes + 255) / 256, 256, 0, stream>>>(lacc, out_lsm, n_nodes);
    // 9) out2 = sigmoid(h @ We^T + be) * att  (bf16 WMMA + fused epilogue)
    k_enc<<<(row_tiles + 7) / 8, 256, 0, stream>>>(hbf, WeT, be, att, out_y, row_tiles);
}
